// Mamba_43009802502633
// MI455X (gfx1250) — matrix operand order repaired and checked by reading, not compiled, not run on hardware
//
#include <hip/hip_runtime.h>
#include <math.h>

// ---------------------------------------------------------------------------
// Mamba-2D vision block for gfx1250 (MI455X, wave32, WMMA).
// All dense contractions use v_wmma_f32_16x16x32_f16 (one 16x16 tile / wave).
// ---------------------------------------------------------------------------

typedef __attribute__((ext_vector_type(16))) _Float16 v16h;
typedef __attribute__((ext_vector_type(8)))  float    v8f;

#define DEV __device__ __forceinline__

namespace {

constexpr int BN = 4;          // batch
constexpr int LL = 4096;       // H*W
constexpr int DM = 96;         // d_model
constexpr int CI = 192;        // d_inner
constexpr int NS = 16;         // d_state
constexpr int RR = 6;          // dt_rank
constexpr int KD = 4;          // scan directions
constexpr int NPROJ = RR + 2 * NS;  // 38

// ------------------------- WMMA 16x16 tile helper --------------------------
// A layout (16-bit, 16x32): lane&15 = row M. halves e:
//   lanes 0-15 : e0-7 -> K=e,     e8-15 -> K=e+8  (i.e. 16..23)
//   lanes16-31 : e0-7 -> K=e+8,   e8-15 -> K=e+16 (i.e. 24..31)
// B layout (16-bit, 32x16): lane&15 = col N. halves e: same K mapping as A
//   lanes 0-15 : e0-7 -> K=e, e8-15 -> K=e+8 ; lanes 16-31 : e0-7 -> K=e+8, e8-15 -> K=e+16
// D layout (f32, 16x16): VGPR r -> M = r + 8*(lane>=16), N = lane&15.
template <typename FA, typename FB>
DEV v8f wmma_mm(int ksteps, FA getA, FB getB) {
  const int lane = threadIdx.x & 31;
  const int hi = lane >> 4;
  v8f acc = {};
  for (int s = 0; s < ksteps; ++s) {
    const int k0 = s << 5;
    v16h av, bv;
#pragma unroll
    for (int e = 0; e < 16; ++e) {
      int ka = k0 + e + ((e & 8) ? 8 : 0) + (hi ? 8 : 0);
      int kb = k0 + e + ((e & 8) ? 8 : 0) + (hi ? 8 : 0);
      av[e] = (_Float16)getA(ka);
      bv[e] = (_Float16)getB(kb);
    }
    acc = __builtin_amdgcn_wmma_f32_16x16x32_f16(
        false, av, false, bv, (short)0, acc, false, false);
  }
  return acc;
}

DEV float silu_f(float v) { return v / (1.f + __expf(-v)); }
DEV float softplus_f(float v) { return v > 20.f ? v : log1pf(__expf(v)); }

// direction index mapping: xs[b,k,c,l] == a[b,c,dirmap(k,l)]
DEV int dirmap(int k, int l) {
  switch (k) {
    case 0:  return l;
    case 1:  return ((l & 63) << 6) | (l >> 6);
    case 2:  return (LL - 1) - l;
    default: { int l2 = (LL - 1) - l; return ((l2 & 63) << 6) | (l2 >> 6); }
  }
}

// ------------------------------ RMSNorm ------------------------------------
__global__ void k_rmsnorm(const float* __restrict__ x,
                          const float* __restrict__ rw,
                          float* __restrict__ xn) {
  const int row = blockIdx.x;           // b*L + l
  const int lane = threadIdx.x;         // 0..31, D = 96 = 3*32
  const float* xr = x + (size_t)row * DM;
  float v0 = xr[lane], v1 = xr[lane + 32], v2 = xr[lane + 64];
  float s = v0 * v0 + v1 * v1 + v2 * v2;
  for (int o = 16; o; o >>= 1) s += __shfl_xor(s, o);
  float inv = rsqrtf(s * (1.f / DM) + 1e-5f);
  float* xo = xn + (size_t)row * DM;
  xo[lane]      = v0 * inv * rw[lane];
  xo[lane + 32] = v1 * inv * rw[lane + 32];
  xo[lane + 64] = v2 * inv * rw[lane + 64];
}

// ------------------------------ in_proj ------------------------------------
// xz = xn @ in_proj_w^T : M=16384, K=96, N=384; cols<192 -> xc (BCHW), else z
__global__ void k_inproj(const float* __restrict__ xn,
                         const float* __restrict__ w,
                         float* __restrict__ xc, float* __restrict__ z) {
  const int tn = blockIdx.x % 24, tm = blockIdx.x / 24;
  const int m0 = tm * 16, n0 = tn * 16;
  const int lane = threadIdx.x & 31, rm = lane & 15, hi = lane >> 4;
  const int arow = m0 + rm, bcol = n0 + rm;
  auto getA = [&](int k) { return xn[(size_t)arow * DM + k]; };
  auto getB = [&](int k) { return w[(size_t)bcol * DM + k]; };
  v8f acc = wmma_mm(3, getA, getB);
#pragma unroll
  for (int r = 0; r < 8; ++r) {
    int m = m0 + r + hi * 8, n = n0 + rm;
    int b = m >> 12, l = m & (LL - 1);
    float v = acc[r];
    if (n < CI) xc[((size_t)b * CI + n) * LL + l] = v;
    else        z[(size_t)m * CI + (n - CI)] = v;
  }
}

// -------------------------------- FFT --------------------------------------
__device__ void fft64(float* re, float* im, float sgn) {
  for (int i = 0; i < 64; ++i) {
    int j = (int)(__brev((unsigned)i) >> 26);
    if (j > i) {
      float t = re[i]; re[i] = re[j]; re[j] = t;
      t = im[i]; im[i] = im[j]; im[j] = t;
    }
  }
  for (int len = 2; len <= 64; len <<= 1) {
    const float ang = sgn * 6.283185307179586f / (float)len;
    const int half = len >> 1;
    for (int i = 0; i < 64; i += len) {
      for (int j = 0; j < half; ++j) {
        float sw, cw;
        __sincosf(ang * (float)j, &sw, &cw);
        float ur = re[i + j], ui = im[i + j];
        float xr = re[i + j + half], xi = im[i + j + half];
        float vr = xr * cw - xi * sw;
        float vi = xr * sw + xi * cw;
        re[i + j] = ur + vr;         im[i + j] = ui + vi;
        re[i + j + half] = ur - vr;  im[i + j + half] = ui - vi;
      }
    }
  }
}

// forward rfft2 (ortho): row rFFT (keep 33 bins) then column FFT.
__global__ void k_fft_fwd(const float* __restrict__ xc,
                          float* __restrict__ xf,         // (img,64,33,2)
                          float* __restrict__ absxf) {    // (img,64,33)
  const int img = blockIdx.x;      // b*C + c
  const int t = threadIdx.x;       // 0..63
  __shared__ float sRe[64 * 33], sIm[64 * 33];
  float re[64], im[64];
  const float* src = xc + (size_t)img * LL;
  for (int w = 0; w < 64; ++w) { re[w] = src[t * 64 + w]; im[w] = 0.f; }
  fft64(re, im, -1.f);
  for (int w = 0; w < 33; ++w) {
    sRe[t * 33 + w] = re[w] * 0.125f;
    sIm[t * 33 + w] = im[w] * 0.125f;
  }
  __syncthreads();
  if (t < 33) {
    for (int h = 0; h < 64; ++h) { re[h] = sRe[h * 33 + t]; im[h] = sIm[h * 33 + t]; }
    fft64(re, im, -1.f);
    for (int h = 0; h < 64; ++h) {
      float rr = re[h] * 0.125f, ii = im[h] * 0.125f;
      size_t o = ((size_t)img * 64 + h) * 33 + t;
      xf[2 * o] = rr; xf[2 * o + 1] = ii;
      absxf[o] = sqrtf(rr * rr + ii * ii);
    }
  }
}

// channel mix: out[b,o,h,w] = act(sum_c in[b,c,h,w] * w[o,c] + bias[o])
__global__ void k_fftmix(const float* __restrict__ in,
                         const float* __restrict__ w,
                         const float* __restrict__ bias,
                         float* __restrict__ out, int do_gelu) {
  const int tn = blockIdx.x % 12, tm = blockIdx.x / 12;   // M=8448, N=192
  const int m0 = tm * 16, n0 = tn * 16;
  const int lane = threadIdx.x & 31, rm = lane & 15, hi = lane >> 4;
  const int arow = m0 + rm, bcol = n0 + rm;
  const int ab = arow / 2112, arem = arow % 2112, ah = arem / 33, aw = arem % 33;
  auto getA = [&](int k) { return in[(((size_t)ab * CI + k) * 64 + ah) * 33 + aw]; };
  auto getB = [&](int k) { return w[(size_t)bcol * CI + k]; };
  v8f acc = wmma_mm(6, getA, getB);
#pragma unroll
  for (int r = 0; r < 8; ++r) {
    int m = m0 + r + hi * 8, n = n0 + rm;
    int b2 = m / 2112, rem2 = m % 2112, h2 = rem2 / 33, w2 = rem2 % 33;
    float v = acc[r] + bias[n];
    if (do_gelu) v = 0.5f * v * (1.f + erff(v * 0.70710678118654752f));
    out[(((size_t)b2 * CI + n) * 64 + h2) * 33 + w2] = v;
  }
}

// inverse: spec = mag * exp(i*angle(xf)); column iFFT, Hermitian row irFFT.
__global__ void k_fft_inv(const float* __restrict__ mag,
                          const float* __restrict__ xf,
                          const float* __restrict__ absxf,
                          float* __restrict__ fourier) {
  const int img = blockIdx.x;
  const int t = threadIdx.x;
  __shared__ float sRe[64 * 33], sIm[64 * 33];
  float re[64], im[64];
  if (t < 33) {
    for (int h = 0; h < 64; ++h) {
      size_t o = ((size_t)img * 64 + h) * 33 + t;
      float av = absxf[o];
      float ur = 1.f, ui = 0.f;
      if (av > 0.f) { ur = xf[2 * o] / av; ui = xf[2 * o + 1] / av; }
      float mg = mag[o];
      re[h] = mg * ur; im[h] = mg * ui;
    }
    fft64(re, im, 1.f);
    for (int h = 0; h < 64; ++h) {
      sRe[h * 33 + t] = re[h] * 0.125f;
      sIm[h * 33 + t] = im[h] * 0.125f;
    }
  }
  __syncthreads();
  for (int w = 0; w < 33; ++w) { re[w] = sRe[t * 33 + w]; im[w] = sIm[t * 33 + w]; }
  for (int w = 33; w < 64; ++w) {
    re[w] = sRe[t * 33 + (64 - w)];
    im[w] = -sIm[t * 33 + (64 - w)];
  }
  fft64(re, im, 1.f);
  float* dst = fourier + (size_t)img * LL + (size_t)t * 64;
  for (int w = 0; w < 64; ++w) dst[w] = re[w] * 0.125f;
}

// ------------------------------ Haar DWT -----------------------------------
__global__ void k_dwt(const float* __restrict__ xc, float* __restrict__ ll,
                      float* __restrict__ lh, float* __restrict__ hl,
                      float* __restrict__ hh) {
  int idx = blockIdx.x * blockDim.x + threadIdx.x;
  if (idx >= BN * CI * 32 * 32) return;
  int x = idx & 31, y = (idx >> 5) & 31, bc = idx >> 10;
  const float* s = xc + (size_t)bc * LL;
  float a = s[(2 * y) * 64 + 2 * x],     b = s[(2 * y) * 64 + 2 * x + 1];
  float c = s[(2 * y + 1) * 64 + 2 * x], d = s[(2 * y + 1) * 64 + 2 * x + 1];
  ll[idx] = (a + b + c + d) * 0.5f;
  lh[idx] = (a - b + c - d) * 0.5f;
  hl[idx] = (a + b - c - d) * 0.5f;
  hh[idx] = (a - b - c + d) * 0.5f;
}

// wavelet convs as implicit GEMM: M = 3*B*32*32 = 12288, K = 192*9, N = 192
__global__ void k_waveconv(const float* __restrict__ lh,
                           const float* __restrict__ hl,
                           const float* __restrict__ hh,
                           const float* __restrict__ w,
                           const float* __restrict__ bias,
                           float* __restrict__ lh2, float* __restrict__ hl2,
                           float* __restrict__ hh2) {
  const int tn = blockIdx.x % 12, tm = blockIdx.x / 12;
  const int m0 = tm * 16, n0 = tn * 16;
  const int lane = threadIdx.x & 31, rm = lane & 15, hi = lane >> 4;
  const int arow = m0 + rm, bcol = n0 + rm;
  const int g = arow >> 12, rem = arow & 4095;
  const int ab = rem >> 10, ay = (rem >> 5) & 31, ax = rem & 31;
  const float* src = (g == 0) ? lh : (g == 1) ? hl : hh;
  auto getA = [&](int k) -> float {
    int cin = k / 9, t = k - cin * 9;
    int yy = ay + t / 3 - 1, xx = ax + t % 3 - 1;
    if ((unsigned)yy < 32u && (unsigned)xx < 32u)
      return src[(((size_t)ab * CI + cin) * 32 + yy) * 32 + xx];
    return 0.f;
  };
  auto getB = [&](int k) { return w[(size_t)bcol * CI * 9 + k]; };
  v8f acc = wmma_mm(54, getA, getB);
#pragma unroll
  for (int r = 0; r < 8; ++r) {
    int m = m0 + r + hi * 8, n = n0 + rm;
    int g2 = m >> 12, rem2 = m & 4095;
    int b2 = rem2 >> 10, y2 = (rem2 >> 5) & 31, x2 = rem2 & 31;
    float v = acc[r] + bias[n];
    float* dst = (g2 == 0) ? lh2 : (g2 == 1) ? hl2 : hh2;
    dst[(((size_t)b2 * CI + n) * 32 + y2) * 32 + x2] = v;
  }
}

// iDWT + residual combine: xc2 = 2*xc + fourier + idwt(ll, lh2, hl2, hh2)
__global__ void k_idwt_comb(const float* __restrict__ xc,
                            const float* __restrict__ fourier,
                            const float* __restrict__ ll,
                            const float* __restrict__ lh2,
                            const float* __restrict__ hl2,
                            const float* __restrict__ hh2,
                            float* __restrict__ xc2) {
  int idx = blockIdx.x * blockDim.x + threadIdx.x;
  if (idx >= BN * CI * 32 * 32) return;
  int x = idx & 31, y = (idx >> 5) & 31, bc = idx >> 10;
  float llv = ll[idx], lhv = lh2[idx], hlv = hl2[idx], hhv = hh2[idx];
  float a = (llv + lhv + hlv + hhv) * 0.5f;
  float b = (llv - lhv + hlv - hhv) * 0.5f;
  float c = (llv + lhv - hlv - hhv) * 0.5f;
  float d = (llv - lhv - hlv + hhv) * 0.5f;
  size_t base = (size_t)bc * LL;
  int p00 = (2 * y) * 64 + 2 * x;
  int p01 = p00 + 1, p10 = p00 + 64, p11 = p00 + 65;
  xc2[base + p00] = 2.f * xc[base + p00] + fourier[base + p00] + a;
  xc2[base + p01] = 2.f * xc[base + p01] + fourier[base + p01] + b;
  xc2[base + p10] = 2.f * xc[base + p10] + fourier[base + p10] + c;
  xc2[base + p11] = 2.f * xc[base + p11] + fourier[base + p11] + d;
}

// ------------------------ depthwise conv + SiLU ----------------------------
__global__ void k_dwconv(const float* __restrict__ xc2,
                         const float* __restrict__ w,
                         const float* __restrict__ bias,
                         float* __restrict__ abuf) {
  int idx = blockIdx.x * blockDim.x + threadIdx.x;
  if (idx >= BN * CI * LL) return;
  int l = idx & (LL - 1), bc = idx >> 12, c = bc % CI;
  int h = l >> 6, x = l & 63;
  const float* src = xc2 + (size_t)bc * LL;
  float s = bias[c];
#pragma unroll
  for (int dy = 0; dy < 3; ++dy)
#pragma unroll
    for (int dx = 0; dx < 3; ++dx) {
      int yy = h + dy - 1, xx = x + dx - 1;
      if ((unsigned)yy < 64u && (unsigned)xx < 64u)
        s += w[c * 9 + dy * 3 + dx] * src[yy * 64 + xx];
    }
  abuf[idx] = silu_f(s);
}

// ------------------------------ x_proj -------------------------------------
// x_dbl[b,k,n,l] = sum_c xs[b,k,c,l] * x_proj_w[k,n,c] ; N=38 padded to 48
__global__ void k_xproj(const float* __restrict__ abuf,
                        const float* __restrict__ xpw,
                        float* __restrict__ xdbl) {
  const int tn = blockIdx.x % 3;
  const int tmp = blockIdx.x / 3;
  const int k = tmp % KD, tm = tmp / KD;
  const int m0 = tm * 16, n0 = tn * 16;
  const int lane = threadIdx.x & 31, rm = lane & 15, hi = lane >> 4;
  const int arow = m0 + rm, bcol = n0 + rm;
  const int ab = arow >> 12, al = arow & (LL - 1);
  const int lm = dirmap(k, al);
  auto getA = [&](int c) { return abuf[((size_t)ab * CI + c) * LL + lm]; };
  auto getB = [&](int c) {
    return (bcol < NPROJ) ? xpw[((size_t)k * NPROJ + bcol) * CI + c] : 0.f;
  };
  v8f acc = wmma_mm(6, getA, getB);
#pragma unroll
  for (int r = 0; r < 8; ++r) {
    int m = m0 + r + hi * 8, n = n0 + rm;
    if (n < NPROJ) {
      int b2 = m >> 12, l2 = m & (LL - 1);
      xdbl[(((size_t)b2 * KD + k) * NPROJ + n) * LL + l2] = acc[r];
    }
  }
}

// --------------------------- dt projection ---------------------------------
// delta[b,k,c,l] = softplus(sum_r dts[b,k,r,l]*dt_w[k,c,r] + dt_b[k,c]); K=6
__global__ void k_dtproj(const float* __restrict__ xdbl,
                         const float* __restrict__ dtw,
                         const float* __restrict__ dtb,
                         float* __restrict__ delta) {
  const int tn = blockIdx.x % 12;
  const int tmp = blockIdx.x / 12;
  const int k = tmp % KD, tm = tmp / KD;
  const int m0 = tm * 16, n0 = tn * 16;
  const int lane = threadIdx.x & 31, rm = lane & 15, hi = lane >> 4;
  const int arow = m0 + rm, bcol = n0 + rm;
  const int ab = arow >> 12, al = arow & (LL - 1);
  auto getA = [&](int r_) {
    return (r_ < RR) ? xdbl[(((size_t)ab * KD + k) * NPROJ + r_) * LL + al] : 0.f;
  };
  auto getB = [&](int r_) {
    return (r_ < RR) ? dtw[((size_t)k * CI + bcol) * RR + r_] : 0.f;
  };
  v8f acc = wmma_mm(1, getA, getB);
#pragma unroll
  for (int r = 0; r < 8; ++r) {
    int m = m0 + r + hi * 8, n = n0 + rm;
    int b2 = m >> 12, l2 = m & (LL - 1);
    float v = softplus_f(acc[r] + dtb[k * CI + n]);
    delta[(((size_t)b2 * KD + k) * CI + n) * LL + l2] = v;
  }
}

// --------------------------- selective scan --------------------------------
__global__ void k_scan(const float* __restrict__ abuf,
                       const float* __restrict__ xdbl,
                       const float* __restrict__ delta,
                       const float* __restrict__ A_logs,
                       const float* __restrict__ Ds,
                       float* __restrict__ oy) {
  int idx = blockIdx.x * blockDim.x + threadIdx.x;   // (b,k,c)
  if (idx >= BN * KD * CI) return;
  const int c = idx % CI, bk = idx / CI;
  const int k = bk % KD, b = bk / KD;
  float Av[NS], h[NS];
#pragma unroll
  for (int n = 0; n < NS; ++n) {
    Av[n] = -__expf(A_logs[((size_t)k * CI + c) * NS + n]);
    h[n] = 0.f;
  }
  const float Dv = Ds[k * CI + c];
  const float* achan = abuf + ((size_t)b * CI + c) * LL;
  const float* drow = delta + (((size_t)b * KD + k) * CI + c) * LL;
  const float* Bbase = xdbl + (((size_t)b * KD + k) * NPROJ + RR) * LL;
  const float* Cbase = xdbl + (((size_t)b * KD + k) * NPROJ + RR + NS) * LL;
  float* orow = oy + (((size_t)b * KD + k) * CI + c) * LL;
  for (int l = 0; l < LL; ++l) {
    float u = achan[dirmap(k, l)];
    float dd = drow[l];
    float du = dd * u;
    float y = 0.f;
#pragma unroll
    for (int n = 0; n < NS; ++n) {
      float Bv = Bbase[(size_t)n * LL + l];
      float Cv = Cbase[(size_t)n * LL + l];
      h[n] = __expf(dd * Av[n]) * h[n] + du * Bv;
      y += h[n] * Cv;
    }
    orow[l] = y + u * Dv;
  }
}

// -------------------- direction merge: y = sum of 4 scans -------------------
__global__ void k_ycomb(const float* __restrict__ oy, float* __restrict__ ybuf) {
  int idx = blockIdx.x * blockDim.x + threadIdx.x;   // (b,c,l)
  if (idx >= BN * CI * LL) return;
  int l = idx & (LL - 1), bc = idx >> 12;
  int c = bc % CI, b = bc / CI;
  int h = l >> 6, w = l & 63;
  int p = (w << 6) | h;
  auto OY = [&](int k, int pos) {
    return oy[(((size_t)b * KD + k) * CI + c) * LL + pos];
  };
  float y = OY(0, l) + OY(2, (LL - 1) - l) + OY(1, p) + OY(3, (LL - 1) - p);
  ybuf[((size_t)b * CI + c) * LL + l] = y;
}

// ------------- symmetry enhance + LayerNorm(C) + SiLU(z) gate --------------
__global__ void k_gate(const float* __restrict__ ybuf,
                       const float* __restrict__ z,
                       const float* __restrict__ symw,
                       const float* __restrict__ g,
                       const float* __restrict__ bb,
                       float* __restrict__ gated) {
  const int row = blockIdx.x;    // b*L + l
  const int lane = threadIdx.x;  // 0..31, C = 192 = 6*32
  const int b = row >> 12, l = row & (LL - 1);
  const int h = l >> 6, w = l & 63;
  const float wgt = 1.f / (1.f + __expf(-symw[0]));
  float vals[6];
  float sum = 0.f;
#pragma unroll
  for (int j = 0; j < 6; ++j) {
    int c = lane + 32 * j;
    size_t base = ((size_t)b * CI + c) * LL;
    float yv = ybuf[base + l];
    float yfl = ybuf[base + (h << 6) + (63 - w)];
    float enh = yv * (1.f - wgt) + 0.5f * (yv + yfl) * wgt;
    vals[j] = yv + 0.2f * enh;
    sum += vals[j];
  }
  for (int o = 16; o; o >>= 1) sum += __shfl_xor(sum, o);
  const float mu = sum * (1.f / CI);
  float vs = 0.f;
#pragma unroll
  for (int j = 0; j < 6; ++j) { float d = vals[j] - mu; vs += d * d; }
  for (int o = 16; o; o >>= 1) vs += __shfl_xor(vs, o);
  const float inv = rsqrtf(vs * (1.f / CI) + 1e-5f);
#pragma unroll
  for (int j = 0; j < 6; ++j) {
    int c = lane + 32 * j;
    float yn = (vals[j] - mu) * inv * g[c] + bb[c];
    float zv = z[(size_t)row * CI + c];
    gated[(size_t)row * CI + c] = yn * silu_f(zv);
  }
}

// ------------------------------ out_proj -----------------------------------
// out = gated @ out_proj_w^T + x : M=16384, K=192, N=96
__global__ void k_outproj(const float* __restrict__ gated,
                          const float* __restrict__ w,
                          const float* __restrict__ xres,
                          float* __restrict__ out) {
  const int tn = blockIdx.x % 6, tm = blockIdx.x / 6;
  const int m0 = tm * 16, n0 = tn * 16;
  const int lane = threadIdx.x & 31, rm = lane & 15, hi = lane >> 4;
  const int arow = m0 + rm, bcol = n0 + rm;
  auto getA = [&](int k) { return gated[(size_t)arow * CI + k]; };
  auto getB = [&](int k) { return w[(size_t)bcol * CI + k]; };
  v8f acc = wmma_mm(6, getA, getB);
#pragma unroll
  for (int r = 0; r < 8; ++r) {
    int m = m0 + r + hi * 8, n = n0 + rm;
    out[(size_t)m * DM + n] = acc[r] + xres[(size_t)m * DM + n];
  }
}

}  // namespace

// ---------------------------------------------------------------------------
extern "C" void kernel_launch(void* const* d_in, const int* in_sizes, int n_in,
                              void* d_out, int out_size, void* d_ws,
                              size_t ws_size, hipStream_t stream) {
  (void)in_sizes; (void)n_in; (void)out_size; (void)ws_size;
  const float* x          = (const float*)d_in[0];
  const float* in_proj_w  = (const float*)d_in[1];
  const float* fft_w1     = (const float*)d_in[2];
  const float* fft_b1     = (const float*)d_in[3];
  const float* fft_w2     = (const float*)d_in[4];
  const float* fft_b2     = (const float*)d_in[5];
  const float* wave_w     = (const float*)d_in[6];
  const float* wave_b     = (const float*)d_in[7];
  const float* dw_w       = (const float*)d_in[8];
  const float* dw_b       = (const float*)d_in[9];
  const float* sym_w      = (const float*)d_in[10];
  const float* x_proj_w   = (const float*)d_in[11];
  const float* dt_w       = (const float*)d_in[12];
  const float* dt_b       = (const float*)d_in[13];
  const float* A_logs     = (const float*)d_in[14];
  const float* Ds         = (const float*)d_in[15];
  const float* out_norm_g = (const float*)d_in[16];
  const float* out_norm_b = (const float*)d_in[17];
  const float* out_proj_w = (const float*)d_in[18];
  const float* rms_w      = (const float*)d_in[19];
  float* out = (float*)d_out;

  // ---- workspace carve (floats); ~65M floats (~260 MB) total ----
  float* W = (float*)d_ws;
  size_t off = 0;
  auto carve = [&](size_t n) { float* p = W + off; off += n; return p; };
  float* xn      = carve((size_t)BN * LL * DM);
  float* zb      = carve((size_t)BN * LL * CI);
  float* xc      = carve((size_t)BN * CI * LL);
  float* xf      = carve((size_t)BN * CI * 64 * 33 * 2);
  float* absxf   = carve((size_t)BN * CI * 64 * 33);
  float* mgelu   = carve((size_t)BN * CI * 64 * 33);
  float* magb    = carve((size_t)BN * CI * 64 * 33);
  float* fourier = carve((size_t)BN * CI * LL);
  float* llb     = carve((size_t)BN * CI * 1024);
  float* lhb     = carve((size_t)BN * CI * 1024);
  float* hlb     = carve((size_t)BN * CI * 1024);
  float* hhb     = carve((size_t)BN * CI * 1024);
  float* lh2     = carve((size_t)BN * CI * 1024);
  float* hl2     = carve((size_t)BN * CI * 1024);
  float* hh2     = carve((size_t)BN * CI * 1024);
  float* xc2     = carve((size_t)BN * CI * LL);
  float* abuf    = carve((size_t)BN * CI * LL);
  float* xdbl    = carve((size_t)BN * KD * NPROJ * LL);
  float* delta   = carve((size_t)BN * KD * CI * LL);
  float* oy      = carve((size_t)BN * KD * CI * LL);
  float* ybuf    = carve((size_t)BN * CI * LL);
  float* gated   = carve((size_t)BN * LL * CI);

  // ---- pipeline ----
  k_rmsnorm<<<BN * LL, 32, 0, stream>>>(x, rms_w, xn);
  k_inproj<<<(BN * LL / 16) * (2 * CI / 16), 32, 0, stream>>>(xn, in_proj_w, xc, zb);
  k_fft_fwd<<<BN * CI, 64, 0, stream>>>(xc, xf, absxf);
  k_fftmix<<<(BN * 64 * 33 / 16) * (CI / 16), 32, 0, stream>>>(absxf, fft_w1, fft_b1, mgelu, 1);
  k_fftmix<<<(BN * 64 * 33 / 16) * (CI / 16), 32, 0, stream>>>(mgelu, fft_w2, fft_b2, magb, 0);
  k_fft_inv<<<BN * CI, 64, 0, stream>>>(magb, xf, absxf, fourier);
  k_dwt<<<(BN * CI * 1024 + 255) / 256, 256, 0, stream>>>(xc, llb, lhb, hlb, hhb);
  k_waveconv<<<(3 * BN * 1024 / 16) * (CI / 16), 32, 0, stream>>>(
      lhb, hlb, hhb, wave_w, wave_b, lh2, hl2, hh2);
  k_idwt_comb<<<(BN * CI * 1024 + 255) / 256, 256, 0, stream>>>(
      xc, fourier, llb, lh2, hl2, hh2, xc2);
  k_dwconv<<<(BN * CI * LL + 255) / 256, 256, 0, stream>>>(xc2, dw_w, dw_b, abuf);
  k_xproj<<<(BN * LL / 16) * KD * 3, 32, 0, stream>>>(abuf, x_proj_w, xdbl);
  k_dtproj<<<(BN * LL / 16) * KD * (CI / 16), 32, 0, stream>>>(xdbl, dt_w, dt_b, delta);
  k_scan<<<(BN * KD * CI + 255) / 256, 256, 0, stream>>>(abuf, xdbl, delta, A_logs, Ds, oy);
  k_ycomb<<<(BN * CI * LL + 255) / 256, 256, 0, stream>>>(oy, ybuf);
  k_gate<<<BN * LL, 32, 0, stream>>>(ybuf, zb, sym_w, out_norm_g, out_norm_b, gated);
  k_outproj<<<(BN * LL / 16) * (DM / 16), 32, 0, stream>>>(gated, out_proj_w, x, out);
}
